// LIIF_3d_43250320671225
// MI455X (gfx1250) — matrix operand order repaired and checked by reading, not compiled, not run on hardware
//
#include <hip/hip_runtime.h>

typedef _Float16 v16h __attribute__((ext_vector_type(16)));
typedef _Float16 h8   __attribute__((ext_vector_type(8)));
typedef float    v8f  __attribute__((ext_vector_type(8)));

#define HW_PIX   61440     // 192*320
#define NROWS    122880    // 2 * HW_PIX
#define ASTR     264       // activation LDS stride in halfs (256 + 8 pad)

// 30 / (2*pi): folds SIREN w0 and v_sin_f32's revolutions-input prescale into one FMA
#define SIREN_SCALE 4.7746482927568600f

// ---- staged-weight GEMM layer: D(16xN) = A(16xK) * W^T + bias, optional sin ----
// ain: per-wave activation rows [16][istr] f16 in LDS
// sw : weights in LDS, row-major [n][K], stride K+8
// outp: SIN -> f16 LDS [16][ASTR];  !SIN -> f32 LDS [16][68]
template<int K, int N, bool SIN>
__device__ __forceinline__ void mm_layer(const _Float16* ain, int istr,
                                         const _Float16* sw,
                                         const float* bias,
                                         void* outp, int lane)
{
    constexpr int KT = K / 32;
    constexpr int NT = N / 16;
    constexpr int WSTR = K + 8;
    const int hk = lane >> 4;    // half-wave group
    const int ln = lane & 15;

    // A fragments: lane ln holds row ln; K-chunks per ISA 16x32 f16 layout
    v16h a[KT];
    const _Float16* arow = ain + ln * istr;
#pragma unroll
    for (int kt = 0; kt < KT; ++kt) {
        h8 lo = *(const h8*)(arow + kt * 32 + hk * 8);
        h8 hi = *(const h8*)(arow + kt * 32 + 16 + hk * 8);
        a[kt] = __builtin_shufflevector(lo, hi, 0,1,2,3,4,5,6,7,8,9,10,11,12,13,14,15);
    }

    for (int nt = 0; nt < NT; ++nt) {
        const int n = nt * 16 + ln;
        const float bv = bias[n];                    // hoisted; hides latency under WMMA
        v8f acc = {0.f, 0.f, 0.f, 0.f, 0.f, 0.f, 0.f, 0.f};  // inline-0 C operand
        const _Float16* wrow = sw + n * WSTR;
#pragma unroll
        for (int kt = 0; kt < KT; ++kt) {
            // B fragment: b[e] = W[n][kt*32 + hk*8 + e] (e<8), W[n][kt*32 + 16 + hk*8 + e-8] (e>=8)
            h8 lo = *(const h8*)(wrow + kt * 32 + hk * 8);
            h8 hi = *(const h8*)(wrow + kt * 32 + 16 + hk * 8);
            v16h b = __builtin_shufflevector(lo, hi, 0,1,2,3,4,5,6,7,8,9,10,11,12,13,14,15);
            acc = __builtin_amdgcn_wmma_f32_16x16x32_f16(
                      false, a[kt], false, b, (short)0, acc, false, false);
        }
        if (SIN) {
            _Float16* ao = (_Float16*)outp;
            const float bb = bv * SIREN_SCALE;       // pre-scaled bias (revolutions)
#pragma unroll
            for (int v = 0; v < 8; ++v) {
                // one v_fma + one v_sin per element; same numerics as __sinf lowering
                float x = __builtin_amdgcn_sinf(__builtin_fmaf(acc[v], SIREN_SCALE, bb));
                ao[(hk * 8 + v) * ASTR + nt * 16 + ln] = (_Float16)x;
            }
        } else {
            float* fo = (float*)outp;
#pragma unroll
            for (int v = 0; v < 8; ++v)
                fo[(hk * 8 + v) * 68 + nt * 16 + ln] = acc[v] + bv;
        }
    }
}

// cooperative stage of one weight matrix (f32 global, [N][SRCK]) -> f16 LDS [n][K], stride K+8
template<int N, int K, int SRCK>
__device__ __forceinline__ void stage_w(_Float16* sw, const float* w, int tid)
{
    for (int idx = tid; idx < N * K; idx += 192) {
        const int n = idx / K;
        const int k = idx - n * K;
        sw[n * (K + 8) + k] = (_Float16)w[n * SRCK + k];
    }
}

__global__ __launch_bounds__(192) void siren_liif_kernel(
    const float* __restrict__ feat, const float* __restrict__ times,
    const float* __restrict__ w0, const float* __restrict__ b0,
    const float* __restrict__ w1, const float* __restrict__ b1,
    const float* __restrict__ w2, const float* __restrict__ b2,
    const float* __restrict__ w3, const float* __restrict__ b3,
    const float* __restrict__ w4, const float* __restrict__ b4,
    const float* __restrict__ w5, const float* __restrict__ b5,
    float* __restrict__ out)
{
    __shared__ _Float16 s_w[256 * 264];        // staged weights (max 256x256, stride 264)
    __shared__ _Float16 s_act0[2 * 16 * 72];   // input feat tiles, 2 rowtiles x 16 rows x 64 (stride 72)
    __shared__ _Float16 s_ping[6 * 16 * ASTR]; // per-wave activation buffers
    __shared__ _Float16 s_pong[6 * 16 * ASTR];
    __shared__ float    s_b0eff[3 * 64];       // effective layer-0 bias per time

    const int tid  = threadIdx.x;
    const int lane = tid & 31;
    const int wv   = tid >> 5;       // 0..5
    const int t    = wv % 3;         // time index for this wave
    const int rt   = wv / 3;         // rowtile (0..1) for this wave

    const int row0 = blockIdx.x * 32;          // 32 consecutive rows per workgroup
    const int bidx = row0 / HW_PIX;            // batch index (constant over wg)
    const int rem0 = row0 - bidx * HW_PIX;     // hw offset within batch

    // ---- stage input activations: feat[(b,c,hw)] -> act0[rowtile][w][c] (f16) ----
    for (int idx = tid; idx < 2048; idx += 192) {
        const int rtt = idx >> 10;
        const int e   = idx & 1023;
        const int k   = e >> 4;                 // channel 0..63
        const int wl  = e & 15;                 // local row
        const float v = feat[(bidx * 64 + k) * HW_PIX + rem0 + rtt * 16 + wl];
        s_act0[rtt * 1152 + wl * 72 + k] = (_Float16)v;
    }
    // ---- stage W0 (drop the t-column: 64x65 -> 64x64) + effective biases ----
    for (int idx = tid; idx < 64 * 64; idx += 192) {
        const int n = idx >> 6, k = idx & 63;
        s_w[n * 72 + k] = (_Float16)w0[n * 65 + k];
    }
    {
        const int tt = tid / 64, n = tid % 64;  // 192 = 3 * 64 exactly
        s_b0eff[tid] = b0[n] + times[tt] * w0[n * 65 + 64];
    }
    __syncthreads();

    _Float16* ping = s_ping + wv * (16 * ASTR);
    _Float16* pong = s_pong + wv * (16 * ASTR);

    // L0: 64 -> 64, eff bias, sin
    mm_layer<64, 64, true>(s_act0 + rt * 1152, 72, s_w, s_b0eff + t * 64, ping, lane);
    __syncthreads();
    stage_w<64, 64, 64>(s_w, w1, tid);
    __syncthreads();
    // L1: 64 -> 64
    mm_layer<64, 64, true>(ping, ASTR, s_w, b1, pong, lane);
    __syncthreads();
    stage_w<256, 64, 64>(s_w, w2, tid);
    __syncthreads();
    // L2: 64 -> 256
    mm_layer<64, 256, true>(pong, ASTR, s_w, b2, ping, lane);
    __syncthreads();
    stage_w<256, 256, 256>(s_w, w3, tid);
    __syncthreads();
    // L3: 256 -> 256
    mm_layer<256, 256, true>(ping, ASTR, s_w, b3, pong, lane);
    __syncthreads();
    stage_w<256, 256, 256>(s_w, w4, tid);
    __syncthreads();
    // L4: 256 -> 256
    mm_layer<256, 256, true>(pong, ASTR, s_w, b4, ping, lane);
    __syncthreads();
    stage_w<64, 256, 256>(s_w, w5, tid);
    __syncthreads();
    // L5: 256 -> 64, linear, f32 result into (reused) pong region
    float* fbuf = (float*)pong;                 // 16 x 68 f32 fits in the 8448B region
    mm_layer<256, 64, false>(ping, ASTR, s_w, b5, fbuf, lane);

    // ---- coalesced store: out[((t*2+b)*64 + n)*HW + hw], contiguous over local row ----
    const int obase = (t * 2 + bidx) * 64;
#pragma unroll 4
    for (int i = 0; i < 32; ++i) {
        const int e = lane + 32 * i;
        const int r = e & 15;
        const int n = e >> 4;
        out[(obase + n) * HW_PIX + rem0 + rt * 16 + r] = fbuf[r * 68 + n];
    }
}

extern "C" void kernel_launch(void* const* d_in, const int* in_sizes, int n_in,
                              void* d_out, int out_size, void* d_ws, size_t ws_size,
                              hipStream_t stream) {
    (void)in_sizes; (void)n_in; (void)out_size; (void)d_ws; (void)ws_size;
    const float* feat  = (const float*)d_in[0];
    const float* times = (const float*)d_in[1];
    const float* w0 = (const float*)d_in[2];  const float* b0 = (const float*)d_in[3];
    const float* w1 = (const float*)d_in[4];  const float* b1 = (const float*)d_in[5];
    const float* w2 = (const float*)d_in[6];  const float* b2 = (const float*)d_in[7];
    const float* w3 = (const float*)d_in[8];  const float* b3 = (const float*)d_in[9];
    const float* w4 = (const float*)d_in[10]; const float* b4 = (const float*)d_in[11];
    const float* w5 = (const float*)d_in[12]; const float* b5 = (const float*)d_in[13];

    dim3 grid(NROWS / 32);   // 3840 workgroups, 32 rows (x3 times) each
    dim3 block(192);         // 6 waves: {rowtile 0,1} x {time 0,1,2}
    siren_liif_kernel<<<grid, block, 0, stream>>>(
        feat, times, w0, b0, w1, b1, w2, b2, w3, b3, w4, b4, w5, b5,
        (float*)d_out);
}
